// lif_snn_47029891891492
// MI455X (gfx1250) — matrix operand order repaired and checked by reading, not compiled, not run on hardware
//
#include <hip/hip_runtime.h>
#include <hip/hip_bf16.h>
#include <stdint.h>

typedef __attribute__((ext_vector_type(16))) _Float16 v16h;
typedef __attribute__((ext_vector_type(8)))  _Float16 v8h;
typedef __attribute__((ext_vector_type(8)))  float    v8f;

#define DECAY  0.951229424500714f   // exp(-1/20)
#define THRESH 0.5f
#define NB 16
#define NT 16

// dense0 split-K parameters
#define D0_KSPLIT 16
#define D0_KRANGE 4096              // K per split (65536/16)
#define D0_STAGE  512               // K staged into LDS per iteration
#define LDS_STRIDE 520              // 512 + 8 halves padding -> conflict-free banks

// ---------------------------------------------------------------------------
// zero the state region (u32 granular, grid-stride)
__global__ void zero_u32_kernel(uint32_t* p, long n) {
    long i  = (long)blockIdx.x * blockDim.x + threadIdx.x;
    long st = (long)gridDim.x * blockDim.x;
    for (; i < n; i += st) p[i] = 0u;
}

// ---------------------------------------------------------------------------
// W1 (OIHW f32, 64x32x3x3) -> f16 transposed to [tap][oc][ic] (tap=ky*3+kx)
__global__ void convert_w1_kernel(const float* __restrict__ W1, _Float16* __restrict__ W1t) {
    int i = blockIdx.x * blockDim.x + threadIdx.x;  // 64*32*9 = 18432
    if (i >= 64 * 32 * 9) return;
    int oc = i / 288, rem = i % 288, ic = rem / 9, tap = rem % 9;
    W1t[tap * 2048 + oc * 32 + ic] = (_Float16)W1[i];
}

// D0 (512x65536 f32) -> f16, same layout (row-major [n][k])
__global__ void convert_d0_kernel(const float* __restrict__ D0, _Float16* __restrict__ D0h, long n) {
    long i  = (long)blockIdx.x * blockDim.x + threadIdx.x;
    long st = (long)gridDim.x * blockDim.x;
    for (; i < n; i += st) D0h[i] = (_Float16)D0[i];
}

// ---------------------------------------------------------------------------
// conv0 (2->32ch, 3x3, pad1) + LIF.  s0/vm0 stored NHWC: idx=((b*64+y)*64+x)*32+oc
__global__ void conv0_lif_kernel(const float* __restrict__ inp, const float* __restrict__ W0,
                                 float* __restrict__ vm0, _Float16* __restrict__ s0h, int t) {
    int idx = blockIdx.x * blockDim.x + threadIdx.x;   // 16*64*64*32 = 2,097,152
    if (idx >= NB * 64 * 64 * 32) return;
    int oc = idx & 31;
    int x  = (idx >> 5)  & 63;
    int y  = (idx >> 11) & 63;
    int b  = idx >> 17;
    const float* ibase = inp + ((long)(b * NT + t) * 2) * 4096;  // (B,T,2,64,64)
    float I = 0.f;
    #pragma unroll
    for (int ic = 0; ic < 2; ++ic)
        #pragma unroll
        for (int ky = 0; ky < 3; ++ky) {
            int yy = y + ky - 1;
            if ((unsigned)yy >= 64u) continue;
            #pragma unroll
            for (int kx = 0; kx < 3; ++kx) {
                int xx = x + kx - 1;
                if ((unsigned)xx >= 64u) continue;
                float v = ibase[ic * 4096 + yy * 64 + xx];
                v = fminf(fmaxf(v, 0.f), 1.f);                 // limit=1 clip
                I += v * W0[((oc * 2 + ic) * 3 + ky) * 3 + kx];
            }
        }
    float sp = (float)s0h[idx];
    float vm = vm0[idx] * DECAY * (1.f - sp) + I;
    vm0[idx] = vm;
    s0h[idx] = (_Float16)(vm > THRESH ? 1.f : 0.f);
}

// ---------------------------------------------------------------------------
// conv1 as implicit-im2col WMMA GEMM: M=B*H*W=65536, N=64, K=9 taps * 32 ic.
// One wave per 16-row M tile; 4 N tiles of 16; 9 K-chunks of 32 -> 36 WMMAs/wave.
__global__ void __launch_bounds__(256)
conv1_wmma_lif_kernel(const _Float16* __restrict__ s0h, const _Float16* __restrict__ W1t,
                      float* __restrict__ vm1, _Float16* __restrict__ s1h) {
    int lane = threadIdx.x & 31;
    int wave = (blockIdx.x * blockDim.x + threadIdx.x) >> 5;   // 0..4095 M tiles
    int r  = lane & 15;
    int hf = lane >> 4;
    int m  = wave * 16 + r;
    int b = m >> 12, pos = m & 4095, y = pos >> 6, x = pos & 63;

    v8f acc[4] = {};
    for (int kk = 0; kk < 9; ++kk) {
        int yy = y + kk / 3 - 1;
        int xx = x + kk % 3 - 1;
        v16h a = {};
        if ((unsigned)yy < 64u && (unsigned)xx < 64u) {
            const _Float16* ap = s0h + ((long)((b * 64 + yy) * 64 + xx)) * 32 + hf * 16;
            v8h lo = *(const v8h*)(ap);
            v8h hi = *(const v8h*)(ap + 8);
            a = __builtin_shufflevector(lo, hi, 0,1,2,3,4,5,6,7,8,9,10,11,12,13,14,15);
        }
        const _Float16* wp = W1t + kk * 2048 + hf * 16;
        #pragma unroll
        for (int nt = 0; nt < 4; ++nt) {
            int oc = nt * 16 + r;
            v16h bb = *(const v16h*)(wp + oc * 32);
            acc[nt] = __builtin_amdgcn_wmma_f32_16x16x32_f16(
                false, a, false, bb, (short)0, acc[nt], false, false);
        }
    }
    // fused LIF epilogue; vm1/s1 are NCHW (B,64,64,64)
    #pragma unroll
    for (int nt = 0; nt < 4; ++nt) {
        int oc = nt * 16 + r;
        #pragma unroll
        for (int i = 0; i < 8; ++i) {
            int m2 = wave * 16 + i + hf * 8;
            int b2 = m2 >> 12, p2 = m2 & 4095, y2 = p2 >> 6, x2 = p2 & 63;
            long idx = ((long)(b2 * 64 + oc)) * 4096 + y2 * 64 + x2;
            float I  = acc[nt][i];
            float sp = (float)s1h[idx];
            float vm = vm1[idx] * DECAY * (1.f - sp) + I;
            vm1[idx] = vm;
            s1h[idx] = (_Float16)(vm > THRESH ? 1.f : 0.f);
        }
    }
}

// ---------------------------------------------------------------------------
// 2x2 maxpool of s1 (NCHW) + flatten -> flat[b][c*1024+py*32+px] (f16)
__global__ void pool_flat_kernel(const _Float16* __restrict__ s1h, _Float16* __restrict__ flat) {
    int i = blockIdx.x * blockDim.x + threadIdx.x;   // 16*64*32*32 = 1,048,576
    if (i >= NB * 64 * 32 * 32) return;
    int px = i & 31, py = (i >> 5) & 31, c = (i >> 10) & 63, b = i >> 16;
    const _Float16* p = s1h + ((long)(b * 64 + c)) * 4096 + (py * 2) * 64 + px * 2;
    float v = fmaxf(fmaxf((float)p[0], (float)p[1]), fmaxf((float)p[64], (float)p[65]));
    flat[(long)b * 65536 + c * 1024 + py * 32 + px] = (_Float16)v;
}

// ---------------------------------------------------------------------------
// dense0 WMMA GEMM, split-K: M=16, N=512 (32 tiles), K=65536 split into 16 ranges.
// Grid: 64 blocks x 256.  blockIdx = {ntile-group(2b) | ksplit(4b)};
// wave w in block -> ntile = group*8 + w.  All 8 waves share one K range, so the
// A operand (16 rows x 512 K per stage) is staged once per block into LDS and
// re-read by every wave via ds_load.  B streams from L2-resident f16 D0 with
// explicit global_prefetch ahead of the WMMA consumer.
__global__ void __launch_bounds__(256)
dense0_wmma_kernel(const _Float16* __restrict__ flat, const _Float16* __restrict__ D0h,
                   float* __restrict__ partial) {
    __shared__ __align__(32) _Float16 As[16 * LDS_STRIDE];
    int tid   = threadIdx.x;
    int lane  = tid & 31;
    int w     = tid >> 5;                    // wave in block: 0..7
    int ks    = blockIdx.x & 15;             // K split: 0..15
    int ntile = (blockIdx.x >> 4) * 8 + w;   // 0..31
    int r  = lane & 15;
    int hf = lane >> 4;
    int n  = ntile * 16 + r;

    const _Float16* brow = D0h + (long)n * 65536 + (long)ks * D0_KRANGE + hf * 16;

    // cooperative A staging: thread tid copies 32 halves of row rr per stage
    int rr = tid >> 4;                       // batch row 0..15
    int cb = (tid & 15) * 32;                // column base within stage
    const _Float16* asrc = flat + (long)rr * 65536 + (long)ks * D0_KRANGE + cb;
    _Float16* adst = As + rr * LDS_STRIDE + cb;

    v8f c = {};
    for (int st = 0; st < D0_KRANGE / D0_STAGE; ++st) {
        #pragma unroll
        for (int j = 0; j < 4; ++j)
            *(v8h*)(adst + j * 8) = *(const v8h*)(asrc + (long)st * D0_STAGE + j * 8);
        __syncthreads();

        const _Float16* arow = As + r * LDS_STRIDE + hf * 16;
        #pragma unroll 4
        for (int c2 = 0; c2 < D0_STAGE / 32; ++c2) {
            int  kb  = c2 * 32;
            long gkb = (long)st * D0_STAGE + kb;
            __builtin_prefetch((const void*)(brow + gkb + 2048), 0, 1);
            v8h lo = *(const v8h*)(arow + kb);
            v8h hi = *(const v8h*)(arow + kb + 8);
            v16h a = __builtin_shufflevector(lo, hi, 0,1,2,3,4,5,6,7,8,9,10,11,12,13,14,15);
            v16h bb = *(const v16h*)(brow + gkb);
            c = __builtin_amdgcn_wmma_f32_16x16x32_f16(
                false, a, false, bb, (short)0, c, false, false);
        }
        __syncthreads();
    }
    // partial[ks][n][b]
    #pragma unroll
    for (int i = 0; i < 8; ++i) {
        int bidx = i + hf * 8;
        partial[((long)ks * 512 + n) * 16 + bidx] = c[i];
    }
}

// reduce the 16 K-split partials + fused LIF on vm2/s2
__global__ void dense0_reduce_lif_kernel(const float* __restrict__ partial,
                                         float* __restrict__ vm2, float* __restrict__ s2) {
    int i = blockIdx.x * blockDim.x + threadIdx.x;   // 512*16 = 8192, i = n*16 + b
    if (i >= 512 * NB) return;
    float I = 0.f;
    #pragma unroll
    for (int ks = 0; ks < D0_KSPLIT; ++ks)
        I += partial[(long)ks * 512 * NB + i];
    int n = i >> 4, b = i & 15;
    int idx = b * 512 + n;
    float sp = s2[idx];
    float vm = vm2[idx] * DECAY * (1.f - sp) + I;
    vm2[idx] = vm;
    s2[idx]  = vm > THRESH ? 1.f : 0.f;
}

// ---------------------------------------------------------------------------
// dense1 (512->11) + LIF + accumulate; writes d_out = acc/16 on last step
__global__ void dense1_lif_acc_kernel(const float* __restrict__ s2, const float* __restrict__ D1,
                                      float* __restrict__ vm3, float* __restrict__ s3,
                                      float* __restrict__ accb, float* __restrict__ out, int last) {
    int i = threadIdx.x;
    if (i >= NB * 11) return;
    int b = i / 11, n = i % 11;
    const float* sp = s2 + b * 512;
    const float* wp = D1 + n * 512;
    float I = 0.f;
    for (int k = 0; k < 512; ++k) I += sp[k] * wp[k];
    float s_prev = s3[i];
    float vm = vm3[i] * DECAY * (1.f - s_prev) + I;
    float s_new = vm > THRESH ? 1.f : 0.f;
    vm3[i] = vm;
    s3[i]  = s_new;
    float a = accb[i] + s_new;
    accb[i] = a;
    if (last) out[i] = a * (1.f / 16.f);
}

// ---------------------------------------------------------------------------
extern "C" void kernel_launch(void* const* d_in, const int* in_sizes, int n_in,
                              void* d_out, int out_size, void* d_ws, size_t ws_size,
                              hipStream_t stream) {
    const float* input = (const float*)d_in[0];   // (16,16,2,64,64)
    const float* W0    = (const float*)d_in[1];   // (32,2,3,3)
    const float* W1    = (const float*)d_in[2];   // (64,32,3,3)
    const float* D0    = (const float*)d_in[3];   // (512,65536)
    const float* D1    = (const float*)d_in[4];   // (11,512)
    float*       out   = (float*)d_out;           // (16,11)

    char*  ws  = (char*)d_ws;
    size_t off = 0;
    auto alloc = [&](size_t bytes) -> char* {
        char* p = ws + off;
        off = (off + bytes + 255) & ~(size_t)255;
        return p;
    };

    // --- state region (zeroed every launch) ---
    float*    vm0  = (float*)alloc((size_t)NB * 32 * 64 * 64 * 4);   // NHWC
    float*    vm1  = (float*)alloc((size_t)NB * 64 * 64 * 64 * 4);   // NCHW
    float*    vm2  = (float*)alloc((size_t)NB * 512 * 4);
    float*    s2   = (float*)alloc((size_t)NB * 512 * 4);
    float*    vm3  = (float*)alloc((size_t)NB * 11 * 4);
    float*    s3   = (float*)alloc((size_t)NB * 11 * 4);
    float*    accb = (float*)alloc((size_t)NB * 11 * 4);
    _Float16* s0h  = (_Float16*)alloc((size_t)NB * 32 * 64 * 64 * 2); // NHWC
    _Float16* s1h  = (_Float16*)alloc((size_t)NB * 64 * 64 * 64 * 2); // NCHW
    _Float16* flat = (_Float16*)alloc((size_t)NB * 65536 * 2);
    size_t stateBytes = off;

    // --- rewritten every launch, no zeroing needed ---
    _Float16* W1t     = (_Float16*)alloc((size_t)9 * 64 * 32 * 2);
    _Float16* D0h     = (_Float16*)alloc((size_t)512 * 65536 * 2);
    float*    partial = (float*)alloc((size_t)D0_KSPLIT * 512 * NB * 4);  // 512 KB

    zero_u32_kernel<<<2048, 256, 0, stream>>>((uint32_t*)ws, (long)(stateBytes / 4));
    convert_w1_kernel<<<(64 * 32 * 9 + 255) / 256, 256, 0, stream>>>(W1, W1t);
    convert_d0_kernel<<<4096, 256, 0, stream>>>(D0, D0h, (long)512 * 65536);

    for (int t = 0; t < NT; ++t) {
        conv0_lif_kernel<<<(NB * 32 * 64 * 64) / 256, 256, 0, stream>>>(input, W0, vm0, s0h, t);
        conv1_wmma_lif_kernel<<<512, 256, 0, stream>>>(s0h, W1t, vm1, s1h);
        pool_flat_kernel<<<(NB * 64 * 32 * 32) / 256, 256, 0, stream>>>(s1h, flat);
        dense0_wmma_kernel<<<64, 256, 0, stream>>>(flat, D0h, partial);
        dense0_reduce_lif_kernel<<<32, 256, 0, stream>>>(partial, vm2, s2);
        dense1_lif_acc_kernel<<<1, 256, 0, stream>>>(s2, D1, vm3, s3, accb, out, t == NT - 1);
    }
    (void)in_sizes; (void)n_in; (void)out_size; (void)ws_size;
}
